// PairformerNoSeqModule_14147622273792
// MI455X (gfx1250) — compile-verified, matrix operand order checked by reading, not run on hardware
//
#include <hip/hip_runtime.h>
#include <math.h>

#define NTOK 256
#define CCH  128
#define HH   4
#define DD   32
#define HD   128
#define MM   (NTOK*NTOK)

typedef __attribute__((ext_vector_type(16))) __bf16 v16bf;
typedef __attribute__((ext_vector_type(8)))  float  v8f;

union FragU { uint4 q[2]; v16bf v; };

__device__ __forceinline__ unsigned short f2bf(float f){
  unsigned u = __builtin_bit_cast(unsigned, f);
  u += 0x7fffu + ((u >> 16) & 1u);           // round-to-nearest-even
  return (unsigned short)(u >> 16);
}
__device__ __forceinline__ float bf2f(unsigned short h){
  unsigned u = ((unsigned)h) << 16;
  return __builtin_bit_cast(float, u);
}
__device__ __forceinline__ float sigf(float x){ return 1.f/(1.f+__expf(-x)); }

// Load one 16x32 bf16 WMMA A/B fragment from row-major [16 rows][ld] storage.
// CDNA5 16-bit A layout: lanes 0-15 -> rows 0-15, K {0..7,16..23};
// lanes 16-31 -> rows 0-15, K {8..15,24..31}. B mirrors it with "row" =
// output column, so the same loader serves both from a [col][K] (W^T) layout.
__device__ __forceinline__ v16bf load_frag(const unsigned short* base, int ld, int k0){
  int lane = threadIdx.x & 31;
  const unsigned short* p = base + (size_t)(lane & 15)*ld + k0 + ((lane >> 4) << 3);
  FragU u;
  u.q[0] = *(const uint4*)(p);
  u.q[1] = *(const uint4*)(p + 16);
  return u.v;
}

__device__ __forceinline__ v8f wmma_bf16(v16bf a, v16bf b, v8f c){
  return __builtin_amdgcn_wmma_f32_16x16x32_bf16(false, a, false, b, (short)0, c, false, false);
}

// ---------------------------------------------------------------- LayerNorm
// One wave per row of 128 channels; fp32 stats, bf16 output.
__global__ void k_layernorm(const float* __restrict__ in,
                            const float* __restrict__ w,
                            const float* __restrict__ b,
                            unsigned short* __restrict__ out){
  int wave = threadIdx.x >> 5, lane = threadIdx.x & 31;
  size_t m = (size_t)blockIdx.x * 8 + wave;
  const float* row = in + m * CCH;
  float4 v4 = *(const float4*)(row + lane * 4);
  float vals[4] = {v4.x, v4.y, v4.z, v4.w};
  float s = vals[0]+vals[1]+vals[2]+vals[3];
  float s2 = vals[0]*vals[0]+vals[1]*vals[1]+vals[2]*vals[2]+vals[3]*vals[3];
  #pragma unroll
  for (int o = 16; o; o >>= 1){ s += __shfl_xor(s, o); s2 += __shfl_xor(s2, o); }
  float mu  = s * (1.f/CCH);
  float var = s2 * (1.f/CCH) - mu*mu;
  float inv = rsqrtf(var + 1e-5f);
  int c = lane * 4;
  #pragma unroll
  for (int i = 0; i < 4; i++)
    out[m*CCH + c + i] = f2bf((vals[i]-mu)*inv*w[c+i] + b[c+i]);
}

// ------------------------------------------------- weight fp32[K,Nc] -> bf16 W^T[Nc,K]
__global__ void k_prep_weight(const float* __restrict__ w,
                              unsigned short* __restrict__ wt, int K, int Nc){
  int t = blockIdx.x * 256 + threadIdx.x;
  if (t >= K*Nc) return;
  int k = t / Nc, n = t - k*Nc;
  wt[(size_t)n*K + k] = f2bf(w[t]);
}

// ---------------------------------------------------------------- GEMM
// out[M,Nc] = A[M,K](bf16) x W (W^T stored [Nc,K] bf16), fp32 accumulate.
// 4-wide N register blocking: one A fragment feeds 4 WMMAs.
// MODE 0: store bf16 (scaled)   MODE 1: store bf16 sigmoid
// MODE 2: outf += acc           MODE 3: outf = acc
template<int MODE>
__global__ void k_gemm(const unsigned short* __restrict__ A,
                       const unsigned short* __restrict__ Wt,
                       int K, int Nc, float scale,
                       float* __restrict__ outf,
                       unsigned short* __restrict__ outh){
  int wave = threadIdx.x >> 5;
  int mt = blockIdx.x * 8 + wave;   // 4096 row tiles
  int nq = blockIdx.y;              // group of 4 column tiles
  const unsigned short* Ab = A  + (size_t)mt*16*K;
  const unsigned short* Bb = Wt + (size_t)nq*64*K;
  v8f acc[4] = {{},{},{},{}};
  for (int k0 = 0; k0 < K; k0 += 32){
    v16bf a = load_frag(Ab, K, k0);
    #pragma unroll
    for (int p = 0; p < 4; p++)
      acc[p] = wmma_bf16(a, load_frag(Bb + (size_t)p*16*K, K, k0), acc[p]);
  }
  int lane = threadIdx.x & 31;
  int mrow = mt*16 + ((lane >> 4) << 3);
  #pragma unroll
  for (int p = 0; p < 4; p++){
    int n = nq*64 + p*16 + (lane & 15);
    #pragma unroll
    for (int r = 0; r < 8; r++){
      size_t idx = (size_t)(mrow + r)*Nc + n;
      float v = acc[p][r];
      if      (MODE == 0) outh[idx] = f2bf(v * scale);
      else if (MODE == 1) outh[idx] = f2bf(sigf(v));
      else if (MODE == 2) outf[idx] += v;
      else                outf[idx] = v;
    }
  }
}

// ---------------------------------------------------------------- dual GEMM
// Two GEMMs sharing the row tile; 2-wide N register blocking.
// MODE 0 (TRIMUL): ab = acc1 * sigmoid(acc2) * mask[m]; scatter channel-major
//                  a_s/b_s (bf16) with optional transpose for incoming edges.
// MODE 1 (GATEOUT): z[m,n] += sigmoid(acc1) * acc2
// MODE 2 (TRANSITION): out16[m,n] = bf16( silu(acc1) * acc2 )
template<int MODE>
__global__ void k_gemm_dual(const unsigned short* __restrict__ A1,
                            const unsigned short* __restrict__ A2,
                            const unsigned short* __restrict__ W1t,
                            const unsigned short* __restrict__ W2t,
                            int K, int Nc,
                            const float* __restrict__ mask, int trans,
                            float* __restrict__ zout,
                            unsigned short* __restrict__ o16a,
                            unsigned short* __restrict__ o16b){
  int wave = threadIdx.x >> 5;
  int mt = blockIdx.x * 8 + wave;
  int np = blockIdx.y;              // pair of column tiles
  const unsigned short* A1b = A1 + (size_t)mt*16*K;
  const unsigned short* A2b = A2 + (size_t)mt*16*K;
  const unsigned short* B1b = W1t + (size_t)np*32*K;
  const unsigned short* B2b = W2t + (size_t)np*32*K;
  v8f acc1[2] = {{},{}}, acc2[2] = {{},{}};
  for (int k0 = 0; k0 < K; k0 += 32){
    v16bf a1 = load_frag(A1b, K, k0);
    v16bf a2 = load_frag(A2b, K, k0);
    #pragma unroll
    for (int p = 0; p < 2; p++){
      acc1[p] = wmma_bf16(a1, load_frag(B1b + (size_t)p*16*K, K, k0), acc1[p]);
      acc2[p] = wmma_bf16(a2, load_frag(B2b + (size_t)p*16*K, K, k0), acc2[p]);
    }
  }
  int lane = threadIdx.x & 31;
  int mrow = mt*16 + ((lane >> 4) << 3);
  #pragma unroll
  for (int p = 0; p < 2; p++){
    int n = np*32 + p*16 + (lane & 15);
    #pragma unroll
    for (int r = 0; r < 8; r++){
      int m = mrow + r;
      if (MODE == 0){
        float val = acc1[p][r] * sigf(acc2[p][r]) * mask[m];
        int c = n & (CCH-1);
        size_t pos = trans ? ((size_t)(m & 255)*NTOK + (m >> 8)) : (size_t)m;
        unsigned short* dst = (n < CCH) ? o16a : o16b;
        dst[(size_t)c*MM + pos] = f2bf(val);
      } else if (MODE == 1){
        zout[(size_t)m*Nc + n] += sigf(acc1[p][r]) * acc2[p][r];
      } else {
        float sx = acc1[p][r];
        o16a[(size_t)m*Nc + n] = f2bf((sx * sigf(sx)) * acc2[p][r]);
      }
    }
  }
}

// ------------------------------------------ triangle einsum: per-channel GEMM
// a_s[c][i][k] (A, row-major), b_s[c][j][k] (B^T, row-major) -> u[i][j][c] fp32
// 4-wide j register blocking: one A fragment feeds 4 WMMAs.
__global__ void k_trimul_einsum(const unsigned short* __restrict__ AS,
                                const unsigned short* __restrict__ BS,
                                float* __restrict__ U){
  int wave = threadIdx.x >> 5;
  int c = blockIdx.y;
  int t = blockIdx.x * 8 + wave;       // 64 tile-groups per channel
  int it = t >> 2, jp = t & 3;
  const unsigned short* Ab = AS + (size_t)c*MM + it*16*NTOK;
  const unsigned short* Bb = BS + (size_t)c*MM + jp*64*NTOK;
  v8f acc[4] = {{},{},{},{}};
  for (int k0 = 0; k0 < NTOK; k0 += 32){
    v16bf a = load_frag(Ab, NTOK, k0);
    #pragma unroll
    for (int p = 0; p < 4; p++)
      acc[p] = wmma_bf16(a, load_frag(Bb + (size_t)p*16*NTOK, NTOK, k0), acc[p]);
  }
  int lane = threadIdx.x & 31;
  int ib = it*16 + ((lane >> 4) << 3);
  #pragma unroll
  for (int p = 0; p < 4; p++){
    int j = jp*64 + p*16 + (lane & 15);
    #pragma unroll
    for (int r = 0; r < 8; r++)
      U[((size_t)(ib + r)*NTOK + j)*CCH + c] = acc[p][r];
  }
}

// --------------------------------------------------- triangle bias projection
__global__ void k_bias_proj(const unsigned short* __restrict__ X,
                            const float* __restrict__ bw,
                            float* __restrict__ tb){
  int t = blockIdx.x * 256 + threadIdx.x;     // M*H
  int h = t & 3; int m = t >> 2;
  const unsigned short* row = X + (size_t)m * CCH;
  float acc = 0.f;
  for (int c = 0; c < CCH; c++) acc += bf2f(row[c]) * bw[c*HH + h];
  tb[t] = acc;
}

// ------------------------------------------------- flash attention per (i,h)
// Q/K/V bf16 [M,HD]; tb fp32 [M,H] bias at (j,k); mask fp32 [N,N] on (i,k).
// out bf16 [M,HD] = gate * softmax_k(qk + bias - (1-mask)*INF) @ V
__global__ void k_attn(const unsigned short* __restrict__ Q,
                       const unsigned short* __restrict__ Kb,
                       const unsigned short* __restrict__ Vb,
                       const float* __restrict__ tb,
                       const float* __restrict__ mask,
                       const unsigned short* __restrict__ gog,
                       unsigned short* __restrict__ out){
  __shared__ unsigned short Ks[NTOK*DD];      // [k][d]
  __shared__ unsigned short Vt[DD*NTOK];      // [d][k]  (transposed)
  __shared__ unsigned short Pl[8*16*32];      // per-wave P tiles
  int i = blockIdx.x, h = blockIdx.y;
  int tid = threadIdx.x;
  { // cooperative stage of K and V slices for this (i,h)
    const unsigned short* kr = Kb + ((size_t)(i*NTOK + tid))*HD + h*DD;
    const unsigned short* vr = Vb + ((size_t)(i*NTOK + tid))*HD + h*DD;
    uint4 k0 = ((const uint4*)kr)[0], k1 = ((const uint4*)kr)[1];
    uint4 k2 = ((const uint4*)kr)[2], k3 = ((const uint4*)kr)[3];
    ((uint4*)(Ks + tid*DD))[0] = k0; ((uint4*)(Ks + tid*DD))[1] = k1;
    ((uint4*)(Ks + tid*DD))[2] = k2; ((uint4*)(Ks + tid*DD))[3] = k3;
    union { uint4 q[4]; unsigned short s[32]; } vv;
    vv.q[0] = ((const uint4*)vr)[0]; vv.q[1] = ((const uint4*)vr)[1];
    vv.q[2] = ((const uint4*)vr)[2]; vv.q[3] = ((const uint4*)vr)[3];
    #pragma unroll
    for (int d = 0; d < 32; d++) Vt[d*NTOK + tid] = vv.s[d];
  }
  __syncthreads();
  int wave = tid >> 5, lane = tid & 31;
  int ln = lane & 15, hl = lane >> 4;
  unsigned short* Pw = Pl + wave*16*32;
  for (int jt = wave; jt < 16; jt += 8){
    int jb = jt * 16;
    v16bf qf = load_frag(Q + ((size_t)(i*NTOK + jb))*HD + h*DD, HD, 0);
    v8f o0 = {}, o1 = {};
    float mrow[8], lrow[8];
    #pragma unroll
    for (int r = 0; r < 8; r++){ mrow[r] = -1e30f; lrow[r] = 0.f; }
    for (int kb = 0; kb < NTOK; kb += 32){
      v8f s0 = {}, s1 = {};
      s0 = wmma_bf16(qf, load_frag(Ks + kb*DD,        DD, 0), s0);
      s1 = wmma_bf16(qf, load_frag(Ks + (kb+16)*DD,   DD, 0), s1);
      int k0c = kb + ln, k1c = kb + 16 + ln;
      float msk0 = (mask[i*NTOK + k0c] - 1.f) * 1e9f;
      float msk1 = (mask[i*NTOK + k1c] - 1.f) * 1e9f;
      #pragma unroll
      for (int r = 0; r < 8; r++){
        int j = jb + r + hl*8;
        float a0 = s0[r] + tb[((size_t)j*NTOK + k0c)*HH + h] + msk0;
        float a1 = s1[r] + tb[((size_t)j*NTOK + k1c)*HH + h] + msk1;
        float cmax = fmaxf(a0, a1);
        #pragma unroll
        for (int o = 8; o; o >>= 1) cmax = fmaxf(cmax, __shfl_xor(cmax, o));
        float mnew = fmaxf(mrow[r], cmax);
        float alpha = __expf(mrow[r] - mnew);
        mrow[r] = mnew;
        float e0 = __expf(a0 - mnew), e1 = __expf(a1 - mnew);
        float rsum = e0 + e1;
        #pragma unroll
        for (int o = 8; o; o >>= 1) rsum += __shfl_xor(rsum, o);
        lrow[r] = lrow[r]*alpha + rsum;
        o0[r] *= alpha; o1[r] *= alpha;
        int prow = r + hl*8;
        Pw[prow*32 + ln]      = f2bf(e0);
        Pw[prow*32 + 16 + ln] = f2bf(e1);
      }
      asm volatile("s_wait_dscnt 0" ::: "memory");   // wave-local LDS RAW
      v16bf pf = load_frag(Pw, 32, 0);
      o0 = wmma_bf16(pf, load_frag(Vt,            NTOK, kb), o0);
      o1 = wmma_bf16(pf, load_frag(Vt + 16*NTOK,  NTOK, kb), o1);
    }
    #pragma unroll
    for (int r = 0; r < 8; r++){
      int j = jb + r + hl*8;
      float invl = 1.f / fmaxf(lrow[r], 1e-20f);
      size_t basei = ((size_t)(i*NTOK + j))*HD + h*DD;
      float g0 = bf2f(gog[basei + ln]);
      float g1 = bf2f(gog[basei + 16 + ln]);
      out[basei + ln]      = f2bf(o0[r]*invl*g0);
      out[basei + 16 + ln] = f2bf(o1[r]*invl*g1);
    }
  }
}

// ---------------------------------------------------------------- transposes
__global__ void k_transpose_z(const float* __restrict__ src, float* __restrict__ dst){
  size_t t = (size_t)blockIdx.x*256 + threadIdx.x;   // M*C
  int c = t & (CCH-1); size_t m = t >> 7;
  int i = (int)(m >> 8), j = (int)(m & 255);
  dst[((size_t)j*NTOK + i)*CCH + c] = src[t];
}
__global__ void k_add_transposed(float* __restrict__ z, const float* __restrict__ dl){
  size_t t = (size_t)blockIdx.x*256 + threadIdx.x;
  int c = t & (CCH-1); size_t m = t >> 7;
  int i = (int)(m >> 8), j = (int)(m & 255);
  z[t] += dl[((size_t)j*NTOK + i)*CCH + c];
}
__global__ void k_transpose_mask(const float* __restrict__ src, float* __restrict__ dst){
  int t = blockIdx.x*256 + threadIdx.x;
  int i = t >> 8, j = t & 255;
  dst[t] = src[j*NTOK + i];
}

// ============================================================ host side
struct Ws {
  float *ZW, *U, *DL, *TB, *MT;
  unsigned short *XB, *XU, *AS, *BS, *VS, *GO, *HB;
  unsigned short *s0, *s1, *s2, *s3, *s4;
};

static void run_trimul(hipStream_t s, const float* mask,
                       const float* ln_w, const float* ln_b,
                       const float* p_w, const float* g_w,
                       const float* ln2_w, const float* ln2_b,
                       const float* out_w, const float* go_w,
                       int outgoing, const Ws& W){
  k_prep_weight<<<(CCH*2*CCH+255)/256, 256, 0, s>>>(p_w,  W.s0, CCH, 2*CCH);
  k_prep_weight<<<(CCH*2*CCH+255)/256, 256, 0, s>>>(g_w,  W.s1, CCH, 2*CCH);
  k_prep_weight<<<(CCH*CCH+255)/256,   256, 0, s>>>(go_w, W.s2, CCH, CCH);
  k_prep_weight<<<(CCH*CCH+255)/256,   256, 0, s>>>(out_w,W.s3, CCH, CCH);
  k_layernorm<<<MM/8, 256, 0, s>>>(W.ZW, ln_w, ln_b, W.XB);
  k_gemm_dual<0><<<dim3(512,8), 256, 0, s>>>(W.XB, W.XB, W.s0, W.s1, CCH, 2*CCH,
                                             mask, outgoing ? 0 : 1,
                                             nullptr, W.AS, W.BS);
  k_trimul_einsum<<<dim3(8,128), 256, 0, s>>>(W.AS, W.BS, W.U);
  k_layernorm<<<MM/8, 256, 0, s>>>(W.U, ln2_w, ln2_b, W.XU);
  k_gemm_dual<1><<<dim3(512,4), 256, 0, s>>>(W.XB, W.XU, W.s2, W.s3, CCH, CCH,
                                             nullptr, 0, W.ZW, nullptr, nullptr);
}

static void run_attn(hipStream_t s, const float* mask_in,
                     const float* ln_w, const float* ln_b,
                     const float* q_w, const float* k_w, const float* v_w,
                     const float* b_w, const float* g_w, const float* o_w,
                     int transposed, const Ws& W){
  const float* maskp = mask_in;
  const float* src = W.ZW;
  if (transposed){
    k_transpose_z<<<MM*CCH/256, 256, 0, s>>>(W.ZW, W.U);
    k_transpose_mask<<<NTOK*NTOK/256, 256, 0, s>>>(mask_in, W.MT);
    src = W.U; maskp = W.MT;
  }
  k_prep_weight<<<(CCH*HD+255)/256, 256, 0, s>>>(q_w, W.s0, CCH, HD);
  k_prep_weight<<<(CCH*HD+255)/256, 256, 0, s>>>(k_w, W.s1, CCH, HD);
  k_prep_weight<<<(CCH*HD+255)/256, 256, 0, s>>>(v_w, W.s2, CCH, HD);
  k_prep_weight<<<(CCH*HD+255)/256, 256, 0, s>>>(g_w, W.s3, CCH, HD);
  k_prep_weight<<<(HD*CCH+255)/256, 256, 0, s>>>(o_w, W.s4, HD, CCH);
  k_layernorm<<<MM/8, 256, 0, s>>>(src, ln_w, ln_b, W.XB);
  const float qscale = 0.17677669529663689f;   // 1/sqrt(32)
  k_gemm<0><<<dim3(512,2), 256, 0, s>>>(W.XB, W.s0, CCH, HD, qscale, nullptr, W.AS);
  k_gemm<0><<<dim3(512,2), 256, 0, s>>>(W.XB, W.s1, CCH, HD, 1.f, nullptr, W.BS);
  k_gemm<0><<<dim3(512,2), 256, 0, s>>>(W.XB, W.s2, CCH, HD, 1.f, nullptr, W.VS);
  k_gemm<1><<<dim3(512,2), 256, 0, s>>>(W.XB, W.s3, CCH, HD, 1.f, nullptr, W.XU);
  k_bias_proj<<<MM*HH/256, 256, 0, s>>>(W.XB, b_w, W.TB);
  k_attn<<<dim3(NTOK,HH), 256, 0, s>>>(W.AS, W.BS, W.VS, W.TB, maskp, W.XU, W.GO);
  if (!transposed){
    k_gemm<2><<<dim3(512,2), 256, 0, s>>>(W.GO, W.s4, HD, CCH, 1.f, W.ZW, nullptr);
  } else {
    k_gemm<3><<<dim3(512,2), 256, 0, s>>>(W.GO, W.s4, HD, CCH, 1.f, W.DL, nullptr);
    k_add_transposed<<<MM*CCH/256, 256, 0, s>>>(W.ZW, W.DL);
  }
}

static void run_transition(hipStream_t s,
                           const float* ln_w, const float* ln_b,
                           const float* fc1, const float* fc2, const float* fc3,
                           const Ws& W){
  k_prep_weight<<<(CCH*4*CCH+255)/256, 256, 0, s>>>(fc1, W.s0, CCH, 4*CCH);
  k_prep_weight<<<(CCH*4*CCH+255)/256, 256, 0, s>>>(fc2, W.s1, CCH, 4*CCH);
  k_prep_weight<<<(4*CCH*CCH+255)/256, 256, 0, s>>>(fc3, W.s2, 4*CCH, CCH);
  k_layernorm<<<MM/8, 256, 0, s>>>(W.ZW, ln_w, ln_b, W.XB);
  k_gemm_dual<2><<<dim3(512,16), 256, 0, s>>>(W.XB, W.XB, W.s0, W.s1, CCH, 4*CCH,
                                              nullptr, 0, nullptr, W.HB, nullptr);
  k_gemm<2><<<dim3(512,2), 256, 0, s>>>(W.HB, W.s2, 4*CCH, CCH, 1.f, W.ZW, nullptr);
}

// flattened input indices (z, pair_mask, then params leaves in sorted-key order)
enum {
  IN_Z = 0, IN_MASK = 1,
  TAE_BW = 2, TAE_GW, TAE_KW, TAE_LNB, TAE_LNW, TAE_OW, TAE_QW, TAE_VW,
  TAS_BW = 10, TAS_GW, TAS_KW, TAS_LNB, TAS_LNW, TAS_OW, TAS_QW, TAS_VW,
  TMI_GW = 18, TMI_GOW, TMI_LN2B, TMI_LN2W, TMI_LNB, TMI_LNW, TMI_OUTW, TMI_PW,
  TMO_GW = 26, TMO_GOW, TMO_LN2B, TMO_LN2W, TMO_LNB, TMO_LNW, TMO_OUTW, TMO_PW,
  TR_FC1 = 34, TR_FC2, TR_FC3, TR_LNB, TR_LNW
};

extern "C" void kernel_launch(void* const* d_in, const int* in_sizes, int n_in,
                              void* d_out, int out_size, void* d_ws, size_t ws_size,
                              hipStream_t stream) {
  (void)in_sizes; (void)n_in; (void)out_size; (void)ws_size;
  unsigned char* ws = (unsigned char*)d_ws;
  const size_t F32 = (size_t)MM*CCH*4, B16 = (size_t)MM*CCH*2;
  Ws W;
  size_t off = 0;
  W.ZW = (float*)(ws + off); off += F32;
  W.U  = (float*)(ws + off); off += F32;
  W.DL = (float*)(ws + off); off += F32;
  W.XB = (unsigned short*)(ws + off); off += B16;
  W.XU = (unsigned short*)(ws + off); off += B16;
  W.AS = (unsigned short*)(ws + off); off += B16;
  W.BS = (unsigned short*)(ws + off); off += B16;
  W.VS = (unsigned short*)(ws + off); off += B16;
  W.GO = (unsigned short*)(ws + off); off += B16;
  W.HB = W.AS;                     // transition hidden [M,512] aliases AS..GO
  W.TB = (float*)(ws + off); off += (size_t)MM*HH*4;
  W.MT = (float*)(ws + off); off += (size_t)NTOK*NTOK*4;
  W.s0 = (unsigned short*)(ws + off); off += 131072;
  W.s1 = (unsigned short*)(ws + off); off += 131072;
  W.s2 = (unsigned short*)(ws + off); off += 131072;
  W.s3 = (unsigned short*)(ws + off); off += 131072;
  W.s4 = (unsigned short*)(ws + off); off += 131072;

  const float* Z0 = (const float*)d_in[IN_Z];
  const float* MK = (const float*)d_in[IN_MASK];
  hipMemcpyAsync(W.ZW, Z0, F32, hipMemcpyDeviceToDevice, stream);

  auto P = [&](int idx, size_t per, int l){ return (const float*)d_in[idx] + (size_t)l*per; };
  const size_t S_C = CCH, S_CH = (size_t)CCH*HD, S_BH = (size_t)CCH*HH;
  const size_t S_C2C = (size_t)CCH*2*CCH, S_CC = (size_t)CCH*CCH, S_C4C = (size_t)CCH*4*CCH;

  for (int l = 0; l < 2; l++){
    run_trimul(stream, MK,
               P(TMO_LNW,S_C,l),  P(TMO_LNB,S_C,l),
               P(TMO_PW,S_C2C,l), P(TMO_GW,S_C2C,l),
               P(TMO_LN2W,S_C,l), P(TMO_LN2B,S_C,l),
               P(TMO_OUTW,S_CC,l),P(TMO_GOW,S_CC,l), 1, W);
    run_trimul(stream, MK,
               P(TMI_LNW,S_C,l),  P(TMI_LNB,S_C,l),
               P(TMI_PW,S_C2C,l), P(TMI_GW,S_C2C,l),
               P(TMI_LN2W,S_C,l), P(TMI_LN2B,S_C,l),
               P(TMI_OUTW,S_CC,l),P(TMI_GOW,S_CC,l), 0, W);
    run_attn(stream, MK,
             P(TAS_LNW,S_C,l), P(TAS_LNB,S_C,l),
             P(TAS_QW,S_CH,l), P(TAS_KW,S_CH,l), P(TAS_VW,S_CH,l),
             P(TAS_BW,S_BH,l), P(TAS_GW,S_CH,l), P(TAS_OW,S_CH,l), 0, W);
    run_attn(stream, MK,
             P(TAE_LNW,S_C,l), P(TAE_LNB,S_C,l),
             P(TAE_QW,S_CH,l), P(TAE_KW,S_CH,l), P(TAE_VW,S_CH,l),
             P(TAE_BW,S_BH,l), P(TAE_GW,S_CH,l), P(TAE_OW,S_CH,l), 1, W);
    run_transition(stream,
                   P(TR_LNW,S_C,l), P(TR_LNB,S_C,l),
                   P(TR_FC1,S_C4C,l), P(TR_FC2,S_C4C,l), P(TR_FC3,S_C4C,l), W);
  }
  hipMemcpyAsync(d_out, W.ZW, F32, hipMemcpyDeviceToDevice, stream);
}
